// Deepseekv3MoE_25013889532221
// MI455X (gfx1250) — matrix operand order repaired and checked by reading, not compiled, not run on hardware
//
#include <hip/hip_runtime.h>
#include <hip/hip_bf16.h>

// ---------------- CDNA5 WMMA types ----------------
typedef __bf16 bf16;
typedef __attribute__((ext_vector_type(16))) __bf16 v16bf;
typedef __attribute__((ext_vector_type(8)))  float  v8f;

#define T_TOK  4096
#define H_DIM  1024
#define I_DIM  256
#define E_EXP  64
#define SI_DIM 1024
#define NGRP   8
#define TOPKG  4
#define TOPK   8
#define RSF    2.5f
#define TM     32           // token tile (2 x 16 WMMA rows)
#define IPAD   (I_DIM + 16) // LDS pad to break bank conflicts

static __device__ __forceinline__ v8f wmma_bf16(v16bf a, v16bf b, v8f c) {
  // v_wmma_f32_16x16x32_bf16: (neg_a, A, neg_b, B, c_mod, C, reuse_a, reuse_b)
  return __builtin_amdgcn_wmma_f32_16x16x32_bf16(false, a, false, b, (short)0, c, false, false);
}

// fast sigmoid: v_exp_f32 + v_rcp_f32 (avoid IEEE div_scale/fma chain)
static __device__ __forceinline__ float fast_sigmoid(float x) {
  return __builtin_amdgcn_rcpf(1.f + __expf(-x));
}

// A fragment 16x32 (MxK), rows contiguous in a bf16 row-major matrix.
// ISA layout: lane m=l&15, hi=l>>4; elem j<8 -> K=j+8*hi ; j>=8 -> K=16+(j-8)+8*hi
static __device__ __forceinline__ v16bf load_a_rows(const bf16* __restrict__ base, int ld,
                                                    int row0, int k0, int lane) {
  int m = lane & 15, hi = lane >> 4;
  const bf16* p = base + (size_t)(row0 + m) * ld + k0 + 8 * hi;
  v16bf f;
#pragma unroll
  for (int j = 0; j < 8; ++j) f[j] = p[j];
#pragma unroll
  for (int j = 0; j < 8; ++j) f[8 + j] = p[16 + j];
  return f;
}

// A fragment with gathered rows (row ids in LDS array), source = global bf16 [.,H_DIM]
static __device__ __forceinline__ v16bf load_a_gather(const bf16* __restrict__ xbf,
                                                      const int* rows, int row0, int k0, int lane) {
  int m = lane & 15, hi = lane >> 4;
  const bf16* p = xbf + (size_t)rows[row0 + m] * H_DIM + k0 + 8 * hi;
  v16bf f;
#pragma unroll
  for (int j = 0; j < 8; ++j) f[j] = p[j];
#pragma unroll
  for (int j = 0; j < 8; ++j) f[8 + j] = p[16 + j];
  return f;
}

// B fragment 32x16 (KxN) from global f32 row-major [K][ldn], converted to bf16.
// ISA layout: lane n=l&15, hi=l>>4; elem j<8 -> K=j+8*hi ; j>=8 -> K=16+(j-8)+8*hi
static __device__ __forceinline__ v16bf load_b_f32(const float* __restrict__ B, int ldn,
                                                   int k0, int n0, int lane) {
  int n = lane & 15, hi = lane >> 4;
  const float* p = B + (size_t)(k0 + 8 * hi) * ldn + n0 + n;
  v16bf f;
#pragma unroll
  for (int j = 0; j < 16; ++j) f[j] = (bf16)p[(size_t)(j < 8 ? j : j + 8) * ldn];
  return f;
}

// prefetch next K-chunk of a weight panel (lowers to global_prefetch_b8)
static __device__ __forceinline__ void prefetch_b(const float* __restrict__ B, int ldn,
                                                  int k0, int n0, int lane) {
  int n = lane & 15, hi = lane >> 4;
  __builtin_prefetch(B + (size_t)(k0 + 16 * hi) * ldn + n0 + n, 0, 1);
}

// ---------------- kernel 1: x -> bf16 ----------------
__global__ void k_cvt_bf16(const float* __restrict__ x, bf16* __restrict__ xbf, int n) {
  int i = blockIdx.x * blockDim.x + threadIdx.x;
  if (i < n) xbf[i] = (bf16)x[i];
}

// ---------------- kernel 2: gate GEMM + noaux_tc routing ----------------
__global__ void k_gate_route(const float* __restrict__ x, const float* __restrict__ gw,
                             const float* __restrict__ bias, float* __restrict__ cw,
                             unsigned char* __restrict__ slotmap) {
  __shared__ float lg[E_EXP];
  int t = blockIdx.x;
  int e = threadIdx.x;
  const float* xr = x + (size_t)t * H_DIM;
  const float* gr = gw + (size_t)e * H_DIM;
  float acc = 0.f;
  for (int h = 0; h < H_DIM; ++h) acc += xr[h] * gr[h];
  lg[e] = acc;
  __syncthreads();
  if (e != 0) return;

  float sc[E_EXP], sb[E_EXP];
  for (int i = 0; i < E_EXP; ++i) {
    float s = fast_sigmoid(lg[i]);
    sc[i] = s;
    sb[i] = s + bias[i];
  }
  // group score = sum of top-2 per group of 8
  float gs[NGRP];
  for (int g = 0; g < NGRP; ++g) {
    float m1 = -1e30f, m2 = -1e30f;
    for (int j = 0; j < E_EXP / NGRP; ++j) {
      float v = sb[g * 8 + j];
      if (v > m1) { m2 = m1; m1 = v; }
      else if (v > m2) { m2 = v; }
    }
    gs[g] = m1 + m2;
  }
  // top-4 groups (strict > keeps lowest index on ties, like jax top_k)
  int gsel[NGRP];
  for (int g = 0; g < NGRP; ++g) gsel[g] = 0;
  for (int k = 0; k < TOPKG; ++k) {
    float best = -1e30f; int bi = 0;
    for (int g = 0; g < NGRP; ++g)
      if (!gsel[g] && gs[g] > best) { best = gs[g]; bi = g; }
    gsel[bi] = 1;
  }
  // mask to selected groups (masked entries become exactly 0, like sb*gmask)
  float sbm[E_EXP];
  for (int i = 0; i < E_EXP; ++i) sbm[i] = gsel[i >> 3] ? sb[i] : 0.0f;
  // top-8 experts
  int sel[E_EXP];
  for (int i = 0; i < E_EXP; ++i) sel[i] = 0;
  for (int k = 0; k < TOPK; ++k) {
    float best = -1e30f; int bi = 0;
    for (int i = 0; i < E_EXP; ++i)
      if (!sel[i] && sbm[i] > best) { best = sbm[i]; bi = i; }
    sel[bi] = 1;
  }
  float sum = 0.f;
  for (int i = 0; i < E_EXP; ++i) if (sel[i]) sum += sc[i];
  float inv = RSF * __builtin_amdgcn_rcpf(sum + 1e-20f);
  float ms[E_EXP];
  for (int i = 0; i < E_EXP; ++i) ms[i] = sel[i] ? sc[i] : 0.0f;

  unsigned char* sm = slotmap + (size_t)t * E_EXP;
  float* cwr = cw + (size_t)t * E_EXP;
  for (int i = 0; i < E_EXP; ++i) sm[i] = 0xFF;
  int used[E_EXP];
  for (int i = 0; i < E_EXP; ++i) used[i] = 0;
  for (int k = 0; k < TOPK; ++k) {
    float best = -1e30f; int bi = 0;
    for (int i = 0; i < E_EXP; ++i)
      if (!used[i] && ms[i] > best) { best = ms[i]; bi = i; }
    used[bi] = 1;
    sm[bi] = (unsigned char)k;          // slot id for this token
    cwr[bi] = ms[bi] * inv;             // routed combine weight
  }
}

// ---------------- kernel 3: deterministic per-expert compaction (1 wave / expert) ----------------
__global__ void k_compact(const float* __restrict__ cw, const unsigned char* __restrict__ slotmap,
                          int* __restrict__ toklist, float* __restrict__ wlist,
                          int* __restrict__ counts) {
  int e = blockIdx.x;
  int lane = threadIdx.x;
  int base = 0;
  for (int t0 = 0; t0 < T_TOK; t0 += 32) {
    int t = t0 + lane;
    unsigned char sl = slotmap[(size_t)t * E_EXP + e];
    bool selv = (sl != 0xFF);
    unsigned long long m = __ballot(selv);
    int off = __popcll(m & ((1ull << lane) - 1ull));
    if (selv) {
      toklist[(size_t)e * T_TOK + base + off] = t | ((int)sl << 16);
      wlist[(size_t)e * T_TOK + base + off]  = cw[(size_t)t * E_EXP + e];
    }
    base += __popcll(m);
  }
  if (lane == 0) counts[e] = base;
}

// ---------------- kernel 4: routed expert FFN (grouped GEMM, WMMA bf16) ----------------
// Grid: (T/TM token tiles, E experts). Block: 256 thr = 8 waves.
// y[t][slot][:] = cw * ( silu(x@w1e) * (x@w3e) ) @ w2e   for gathered tokens.
__global__ void k_routed_ffn(const bf16* __restrict__ xbf,
                             const float* __restrict__ w1, const float* __restrict__ w3,
                             const float* __restrict__ w2,
                             const int* __restrict__ toklist, const float* __restrict__ wlist,
                             const int* __restrict__ counts, float* __restrict__ ybuf) {
  int e = blockIdx.y;
  int tile = blockIdx.x;
  int cnt = counts[e];
  if (tile * TM >= cnt) return;   // uniform exit: EXEC stays all-1s for WMMA

  __shared__ int   s_tok[TM];
  __shared__ int   s_slot[TM];
  __shared__ float s_w[TM];
  __shared__ bf16  htile[TM][IPAD];

  int tid = threadIdx.x;
  int lane = tid & 31, wv = tid >> 5;
  if (tid < TM) {
    int r = tile * TM + tid;
    if (r < cnt) {
      int p = toklist[(size_t)e * T_TOK + r];
      s_tok[tid] = p & 0xFFFF;
      s_slot[tid] = p >> 16;
      s_w[tid] = wlist[(size_t)e * T_TOK + r];
    } else {
      s_tok[tid] = 0; s_slot[tid] = 0; s_w[tid] = 0.f;
    }
  }
  __syncthreads();

  const float* W1 = w1 + (size_t)e * H_DIM * I_DIM;
  const float* W3 = w3 + (size_t)e * H_DIM * I_DIM;
  const float* W2 = w2 + (size_t)e * I_DIM * H_DIM;

  int mt = wv & 1, ng = wv >> 1;          // 2 m-tiles x 4 wave-groups
  int n = lane & 15, hi = lane >> 4;

  // ---- phase A: h[32 x 256] = silu(X@W1) * (X@W3), 4 n-tiles per wave ----
  for (int c = 0; c < 4; ++c) {
    int n0 = (ng * 4 + c) * 16;
    v8f a1 = {}, a3 = {};
    for (int k0 = 0; k0 < H_DIM; k0 += 32) {
      v16bf A  = load_a_gather(xbf, s_tok, 16 * mt, k0, lane);
      v16bf B1 = load_b_f32(W1, I_DIM, k0, n0, lane);
      v16bf B3 = load_b_f32(W3, I_DIM, k0, n0, lane);
      if (k0 + 32 < H_DIM) {   // uniform guard; hide next chunk's strided-load latency
        prefetch_b(W1, I_DIM, k0 + 32, n0, lane);
        prefetch_b(W3, I_DIM, k0 + 32, n0, lane);
      }
      a1 = wmma_bf16(A, B1, a1);
      a3 = wmma_bf16(A, B3, a3);
    }
#pragma unroll
    for (int r = 0; r < 8; ++r) {
      float v1 = a1[r], v3 = a3[r];
      float hv = v1 * fast_sigmoid(v1) * v3;        // silu(v1) * v3
      htile[16 * mt + r + 8 * hi][n0 + n] = (bf16)hv;
    }
  }
  __syncthreads();

  // ---- phase B: y[32 x 1024] = h @ W2, weighted scatter into ybuf ----
  for (int c = 0; c < 16; ++c) {
    int n0 = (ng * 16 + c) * 16;
    v8f acc = {};
    for (int k0 = 0; k0 < I_DIM; k0 += 32) {
      v16bf A = load_a_rows(&htile[0][0], IPAD, 16 * mt, k0, lane);
      v16bf B = load_b_f32(W2, H_DIM, k0, n0, lane);
      if (k0 + 32 < I_DIM) prefetch_b(W2, H_DIM, k0 + 32, n0, lane);
      acc = wmma_bf16(A, B, acc);
    }
#pragma unroll
    for (int r = 0; r < 8; ++r) {
      int m = 16 * mt + r + 8 * hi;
      if (tile * TM + m < cnt) {
        int t = s_tok[m], sl = s_slot[m];
        ybuf[((size_t)t * TOPK + sl) * H_DIM + n0 + n] = acc[r] * s_w[m];
      }
    }
  }
}

// ---------------- kernel 5: shared expert phase A: h = silu(x@sw1)*(x@sw3) ----------------
__global__ void k_shared_a(const bf16* __restrict__ xbf, const float* __restrict__ sw1,
                           const float* __restrict__ sw3, bf16* __restrict__ hbuf) {
  int t0 = blockIdx.x * TM;
  int lane = threadIdx.x & 31, wv = threadIdx.x >> 5;
  int mt = wv & 1, ng = wv >> 1;
  int n = lane & 15, hi = lane >> 4;
  for (int c = 0; c < 16; ++c) {
    int n0 = (ng * 16 + c) * 16;
    v8f a1 = {}, a3 = {};
    for (int k0 = 0; k0 < H_DIM; k0 += 32) {
      v16bf A  = load_a_rows(xbf, H_DIM, t0 + 16 * mt, k0, lane);
      v16bf B1 = load_b_f32(sw1, SI_DIM, k0, n0, lane);
      v16bf B3 = load_b_f32(sw3, SI_DIM, k0, n0, lane);
      if (k0 + 32 < H_DIM) {
        prefetch_b(sw1, SI_DIM, k0 + 32, n0, lane);
        prefetch_b(sw3, SI_DIM, k0 + 32, n0, lane);
      }
      a1 = wmma_bf16(A, B1, a1);
      a3 = wmma_bf16(A, B3, a3);
    }
#pragma unroll
    for (int r = 0; r < 8; ++r) {
      float v1 = a1[r], v3 = a3[r];
      float hv = v1 * fast_sigmoid(v1) * v3;
      hbuf[(size_t)(t0 + 16 * mt + r + 8 * hi) * SI_DIM + n0 + n] = (bf16)hv;
    }
  }
}

// ---------------- kernel 6: shared expert phase B: out = h @ sw2 ----------------
__global__ void k_shared_b(const bf16* __restrict__ hbuf, const float* __restrict__ sw2,
                           float* __restrict__ out) {
  int t0 = blockIdx.x * TM;
  int lane = threadIdx.x & 31, wv = threadIdx.x >> 5;
  int mt = wv & 1, ng = wv >> 1;
  int n = lane & 15, hi = lane >> 4;
  for (int c = 0; c < 16; ++c) {
    int n0 = (ng * 16 + c) * 16;
    v8f acc = {};
    for (int k0 = 0; k0 < SI_DIM; k0 += 32) {
      v16bf A = load_a_rows(hbuf, SI_DIM, t0 + 16 * mt, k0, lane);
      v16bf B = load_b_f32(sw2, H_DIM, k0, n0, lane);
      if (k0 + 32 < SI_DIM) prefetch_b(sw2, H_DIM, k0 + 32, n0, lane);
      acc = wmma_bf16(A, B, acc);
    }
#pragma unroll
    for (int r = 0; r < 8; ++r)
      out[(size_t)(t0 + 16 * mt + r + 8 * hi) * H_DIM + n0 + n] = acc[r];
  }
}

// ---------------- kernel 7: out += sum over 8 slots of ybuf (deterministic combine) ----------------
__global__ void k_reduce(const float* __restrict__ ybuf, float* __restrict__ out) {
  int i4 = blockIdx.x * blockDim.x + threadIdx.x;
  const int H4 = H_DIM / 4;
  if (i4 >= T_TOK * H4) return;
  int t = i4 / H4, h4 = i4 % H4;
  float4 acc = reinterpret_cast<float4*>(out)[i4];
  const float4* yb = reinterpret_cast<const float4*>(ybuf);
#pragma unroll
  for (int s = 0; s < TOPK; ++s) {
    float4 y = yb[(size_t)(t * TOPK + s) * H4 + h4];
    acc.x += y.x; acc.y += y.y; acc.z += y.z; acc.w += y.w;
  }
  reinterpret_cast<float4*>(out)[i4] = acc;
}

// ---------------- host launch ----------------
extern "C" void kernel_launch(void* const* d_in, const int* in_sizes, int n_in,
                              void* d_out, int out_size, void* d_ws, size_t ws_size,
                              hipStream_t stream) {
  (void)in_sizes; (void)n_in; (void)out_size; (void)ws_size;
  const float* x    = (const float*)d_in[0];
  const float* gw   = (const float*)d_in[1];
  const float* bias = (const float*)d_in[2];
  const float* w1   = (const float*)d_in[3];
  const float* w3   = (const float*)d_in[4];
  const float* w2   = (const float*)d_in[5];
  const float* sw1  = (const float*)d_in[6];
  const float* sw3  = (const float*)d_in[7];
  const float* sw2  = (const float*)d_in[8];
  float* out = (float*)d_out;

  // workspace carve-up (all offsets 256B aligned)
  char* ws = (char*)d_ws;
  size_t off = 0;
  bf16* xbf = (bf16*)(ws + off);              off += (size_t)T_TOK * H_DIM * sizeof(bf16);   // 8 MB
  float* cw = (float*)(ws + off);             off += (size_t)T_TOK * E_EXP * sizeof(float);  // 1 MB
  unsigned char* slotmap = (unsigned char*)(ws + off); off += (size_t)T_TOK * E_EXP;         // 256 KB
  int* counts = (int*)(ws + off);             off += 256;
  int* toklist = (int*)(ws + off);            off += (size_t)E_EXP * T_TOK * sizeof(int);    // 1 MB
  float* wlist = (float*)(ws + off);          off += (size_t)E_EXP * T_TOK * sizeof(float);  // 1 MB
  bf16* hbuf = (bf16*)(ws + off);             off += (size_t)T_TOK * SI_DIM * sizeof(bf16);  // 8 MB
  float* ybuf = (float*)(ws + off);           off += (size_t)T_TOK * TOPK * H_DIM * sizeof(float); // 128 MB

  k_cvt_bf16<<<(T_TOK * H_DIM + 255) / 256, 256, 0, stream>>>(x, xbf, T_TOK * H_DIM);
  k_gate_route<<<T_TOK, E_EXP, 0, stream>>>(x, gw, bias, cw, slotmap);
  k_compact<<<E_EXP, 32, 0, stream>>>(cw, slotmap, toklist, wlist, counts);
  k_routed_ffn<<<dim3(T_TOK / TM, E_EXP), 256, 0, stream>>>(xbf, w1, w3, w2, toklist, wlist,
                                                            counts, ybuf);
  k_shared_a<<<T_TOK / TM, 256, 0, stream>>>(xbf, sw1, sw3, hbuf);
  k_shared_b<<<T_TOK / TM, 256, 0, stream>>>(hbuf, sw2, out);
  k_reduce<<<(T_TOK * (H_DIM / 4) + 255) / 256, 256, 0, stream>>>(ybuf, out);
}
